// MODEL_57157424775519
// MI455X (gfx1250) — matrix operand order repaired and checked by reading, not compiled, not run on hardware
//
#include <hip/hip_runtime.h>
#include <hip/hip_bf16.h>
#include <math.h>

// ---------------------------------------------------------------------------
// DKVMN-style knowledge-tracing forward pass for MI455X (gfx1250).
//
// Design (see analysis): all dense GEMMs run as wave32 WMMA f16->f32
// (v_wmma_f32_16x16x32_f16), one 16x16 output tile per wave, activations and
// weights packed to zero-padded f16 (D=200 -> 224, 2D=400 -> 416).  The
// erase/add GEMMs are hoisted out of the T=200 recurrence (they only depend
// on qa), so the scan itself is pure VALU with the [50,200] memory state
// resident in LDS (40 KB) — one workgroup per batch element (128 WGs).
// ---------------------------------------------------------------------------

typedef __attribute__((ext_vector_type(16))) _Float16 v16h;
typedef __attribute__((ext_vector_type(8)))  _Float16 v8h;
typedef __attribute__((ext_vector_type(8)))  float    v8f;

#define B_   128
#define T_   200
#define D_   200
#define DP   224          // D padded: multiple of 32 (K) and 16 (N)
#define MEM_ 50
#define FC_  512
#define M_   (B_ * T_)    // 25600 rows
#define P2   416          // 2*D padded to multiple of 32

// ---------------------------------------------------------------------------
// Generic WMMA GEMM:  C[M,N] = act( A[M,Kp] @ W[N,Kp]^T + bias )
// A, W are f16 with row stride Kp (zero-padded). One wave -> one 16x16 tile.
// Fragment layouts per CDNA5 ISA 7.12.2:
//   A (16x32 f16): lane<16 holds row (lane), halves K=[0..7]+[16..23];
//                  lane>=16 holds row (lane-16), halves K=[8..15]+[24..31].
//   B (32x16 f16): lane holds column (lane&15); lane<16 halves K=[0..7]+[16..23],
//                  lane>=16 halves K=[8..15]+[24..31].
//   C/D (16x16 f32): acc[r] = C[mBase + r + (lane>=16?8:0)][nBase + (lane&15)]
// Each half-fragment chunk is 8 contiguous halves = one 16B global load.
// ---------------------------------------------------------------------------
__global__ __launch_bounds__(256) void gemm_wmma(
    const _Float16* __restrict__ A, const _Float16* __restrict__ W,
    const float* __restrict__ bias, int Nreal,
    float* __restrict__ Cf, _Float16* __restrict__ Ch,
    int Kp, int nTilesN, int totalTiles, int ldC, int act)
{
    int wave = threadIdx.x >> 5;
    int lane = threadIdx.x & 31;
    int tt = blockIdx.x * 8 + wave;
    if (tt >= totalTiles) return;                 // wave-uniform, EXEC stays full

    int mTile = tt / nTilesN;
    int nTile = tt - mTile * nTilesN;
    int mBase = mTile * 16, nBase = nTile * 16;
    int hl = lane & 15;
    bool hi = lane >= 16;

    const _Float16* Arow = A + (size_t)(mBase + hl) * Kp;
    const _Float16* Wrow = W + (size_t)(nBase + hl) * Kp;
    const int aLo = hi ? 8  : 0,  aHi = hi ? 24 : 16;
    const int bLo = hi ? 8  : 0,  bHi = hi ? 24 : 16;

    v8f acc = {};
    for (int k0 = 0; k0 < Kp; k0 += 32) {
        __builtin_prefetch(Arow + k0 + 32, 0, 1);  // global_prefetch_b8
        __builtin_prefetch(Wrow + k0 + 32, 0, 1);
        v8h a0 = *(const v8h*)(Arow + k0 + aLo);
        v8h a1 = *(const v8h*)(Arow + k0 + aHi);
        v8h b0 = *(const v8h*)(Wrow + k0 + bLo);
        v8h b1 = *(const v8h*)(Wrow + k0 + bHi);
        v16h af = __builtin_shufflevector(a0, a1, 0,1,2,3,4,5,6,7,8,9,10,11,12,13,14,15);
        v16h bf = __builtin_shufflevector(b0, b1, 0,1,2,3,4,5,6,7,8,9,10,11,12,13,14,15);
        acc = __builtin_amdgcn_wmma_f32_16x16x32_f16(
                false, af, false, bf, (short)0, acc, false, false);
    }

    int col = nBase + hl;
    float bv = (col < Nreal) ? bias[col] : 0.f;
#pragma unroll
    for (int r = 0; r < 8; r++) {
        int row = mBase + r + (hi ? 8 : 0);
        float v = acc[r] + bv;
        if      (act == 1) v = tanhf(v);
        else if (act == 2) v = 1.f / (1.f + __expf(-v));
        if (Cf) Cf[(size_t)row * ldC + col] = v;
        if (Ch) Ch[(size_t)row * ldC + col] = (_Float16)v;
    }
}

// ---------------------------------------------------------------------------
// Pack fp32 weight [N,K] -> zero-padded f16 [Nrows,Kp]
// ---------------------------------------------------------------------------
__global__ void pack_weight(const float* __restrict__ W, _Float16* __restrict__ Wh,
                            int N, int K, int Kp, int Nrows)
{
    int idx = blockIdx.x * 256 + threadIdx.x;
    if (idx >= Nrows * Kp) return;
    int n = idx / Kp, k = idx - n * Kp;
    float v = (n < N && k < K) ? W[(size_t)n * K + k] : 0.f;
    Wh[idx] = (_Float16)v;
}

// ---------------------------------------------------------------------------
// Embedding gathers + packing of the pre-scan GEMM operands.
//   HA  = f16(t_e + q_e)        (input of diff path for time_final)
//   HB  = f16(ht_e + h_e + a_e) (ctx, input of diff path for diff_final)
//   HQA = f16(qa_e)             (input of erase/add GEMMs)
// fp32 copies kept for later elementwise math (stride DP, pad = 0).
// ---------------------------------------------------------------------------
__global__ __launch_bounds__(224) void embed_pack(
    const int* __restrict__ qd,  const int* __restrict__ qad,
    const int* __restrict__ td,  const int* __restrict__ ad,
    const int* __restrict__ hd,  const int* __restrict__ htd,
    const float* __restrict__ qE,  const float* __restrict__ qaE,
    const float* __restrict__ tE,  const float* __restrict__ aE,
    const float* __restrict__ htE,
    _Float16* __restrict__ HA, _Float16* __restrict__ HB, _Float16* __restrict__ HQA,
    float* __restrict__ Fqe, float* __restrict__ Fte, float* __restrict__ Fctx)
{
    size_t row = blockIdx.x;
    int d = threadIdx.x;                      // 0..223
    float qv = 0.f, tv = 0.f, av = 0.f, hv = 0.f, htv = 0.f, qav = 0.f;
    if (d < D_) {
        int qi  = qd[row],  qai = qad[row], ti = td[row];
        int ai  = ad[row],  hi_ = hd[row],  hti = htd[row];
        qv  = qE [(size_t)qi  * D_ + d];
        qav = qaE[(size_t)qai * D_ + d];
        tv  = tE [(size_t)ti  * D_ + d];
        av  = aE [(size_t)ai  * D_ + d];
        hv  = htE[(size_t)hi_ * D_ + d];   // forward uses hint_total_embed for hint_data
        htv = htE[(size_t)hti * D_ + d];
    }
    float ctx = htv + hv + av;
    HA [row * DP + d] = (_Float16)(tv + qv);
    HB [row * DP + d] = (_Float16)ctx;
    HQA[row * DP + d] = (_Float16)qav;
    Fqe[row * DP + d] = qv;
    Fte[row * DP + d] = tv;
    Fctx[row * DP + d] = ctx;
}

// Y = q_e + diff_final * ctx  -> f16 (reuses HA)
__global__ __launch_bounds__(224) void combine_y(
    const float* __restrict__ Fqe, const float* __restrict__ Fdf,
    const float* __restrict__ Fctx, _Float16* __restrict__ HY)
{
    size_t row = blockIdx.x;
    int d = threadIdx.x;
    float v = 0.f;
    if (d < D_) v = Fqe[row * DP + d] + Fdf[row * DP + d] * Fctx[row * DP + d];
    HY[row * DP + d] = (_Float16)v;
}

// ---------------------------------------------------------------------------
// Recurrent memory scan. One workgroup per batch element; the [50,200] value
// memory lives in LDS (40 KB) for all 200 steps. erase/add were precomputed
// by WMMA GEMMs (Fe/Fa). Writes key/value reads as packed f16 rows.
// ---------------------------------------------------------------------------
__global__ __launch_bounds__(256) void scan_kernel(
    const float* __restrict__ Fqf, const float* __restrict__ Fe,
    const float* __restrict__ Fa,  const float* __restrict__ memKey,
    const float* __restrict__ memV0,
    _Float16* __restrict__ HR, _Float16* __restrict__ HVR)
{
    __shared__ float sMv[MEM_ * D_];     // 40 KB value memory
    __shared__ float sQ[D_];
    __shared__ float sLc[MEM_], sLv[MEM_], sCw[MEM_], sVw[MEM_];

    int b = blockIdx.x, tid = threadIdx.x;
    for (int i = tid; i < MEM_ * D_; i += 256) sMv[i] = memV0[i];
    __syncthreads();

    for (int t = 0; t < T_; t++) {
        size_t row = (size_t)b * T_ + t;
        if (tid < D_) sQ[tid] = Fqf[row * DP + tid];
        __syncthreads();

        if (tid < MEM_) {                                  // key attention logits
            float s = 0.f;
            for (int d = 0; d < D_; d++) s += sQ[d] * memKey[tid * D_ + d];
            sLc[tid] = s;
        } else if (tid >= 64 && tid < 64 + MEM_) {         // value attention logits
            int m = tid - 64;
            float s = 0.f;
            for (int d = 0; d < D_; d++) s += sQ[d] * sMv[m * D_ + d];
            sLv[m] = s;
        }
        __syncthreads();

        if (tid == 0) {                                    // softmax over 50 (key)
            float mx = -1e30f;
            for (int m = 0; m < MEM_; m++) mx = fmaxf(mx, sLc[m]);
            float s = 0.f;
            for (int m = 0; m < MEM_; m++) { float e = __expf(sLc[m] - mx); sCw[m] = e; s += e; }
            float inv = 1.f / s;
            for (int m = 0; m < MEM_; m++) sCw[m] *= inv;
        } else if (tid == 32) {                            // softmax over 50 (value)
            float mx = -1e30f;
            for (int m = 0; m < MEM_; m++) mx = fmaxf(mx, sLv[m]);
            float s = 0.f;
            for (int m = 0; m < MEM_; m++) { float e = __expf(sLv[m] - mx); sVw[m] = e; s += e; }
            float inv = 1.f / s;
            for (int m = 0; m < MEM_; m++) sVw[m] *= inv;
        }
        __syncthreads();

        if (tid < D_) {                                    // reads (pre-update Mv)
            float r = 0.f, vr = 0.f;
            for (int m = 0; m < MEM_; m++) {
                float mv = sMv[m * D_ + tid];
                r  += sCw[m] * mv;
                vr += sVw[m] * mv;
            }
            HR [row * DP + tid] = (_Float16)r;
            HVR[row * DP + tid] = (_Float16)vr;
        } else if (tid < DP) {                             // keep pad columns zero
            HR [row * DP + tid] = (_Float16)0.f;
            HVR[row * DP + tid] = (_Float16)0.f;
        }
        __syncthreads();

        const float* Er = Fe + row * DP;                   // memory update
        const float* Ar = Fa + row * DP;
        for (int i = tid; i < MEM_ * D_; i += 256) {
            int m = i / D_, d = i - m * D_;
            float c = sCw[m];
            sMv[i] = sMv[i] * (1.f - c * Er[d]) + c * Ar[d];
        }
        __syncthreads();
    }
}

// ---------------------------------------------------------------------------
// Joint LayerNorm over [T,D] per batch + sigmoid gate + predict_input pack.
// HP row = [ output_weight(200) | input_embed(200) | zeros(16) ]  as f16.
// ---------------------------------------------------------------------------
__global__ __launch_bounds__(256) void ln_predictpack(
    const float* __restrict__ Frv,  const float* __restrict__ Frvk,
    const float* __restrict__ Fqf,  const float* __restrict__ Ftf,
    const float* __restrict__ Fte,  const float* __restrict__ gamma,
    const float* __restrict__ beta, _Float16* __restrict__ HP)
{
    __shared__ float red[256], red2[256];
    __shared__ float sMu, sRs;
    int b = blockIdx.x, tid = threadIdx.x;
    const int N = T_ * D_;

    float s = 0.f, s2 = 0.f;
    for (int i = tid; i < N; i += 256) {
        int t = i / D_, d = i - t * D_;
        float v = Frvk[((size_t)b * T_ + t) * DP + d];
        s += v; s2 += v * v;
    }
    red[tid] = s; red2[tid] = s2;
    __syncthreads();
    for (int o = 128; o > 0; o >>= 1) {
        if (tid < o) { red[tid] += red[tid + o]; red2[tid] += red2[tid + o]; }
        __syncthreads();
    }
    if (tid == 0) {
        float mu = red[0] / (float)N;
        float var = red2[0] / (float)N - mu * mu;
        sMu = mu;
        sRs = rsqrtf(var + 1e-5f);
    }
    __syncthreads();
    float mu = sMu, rs = sRs;

    for (int i = tid; i < N; i += 256) {
        int t = i / D_, d = i - t * D_;
        size_t row = (size_t)b * T_ + t;
        float v  = Frvk[row * DP + d];
        float vn = (v - mu) * rs * gamma[i] + beta[i];
        float ow = 1.f / (1.f + __expf(-(Frv[row * DP + d] + vn)));
        float ie = Fqf[row * DP + d] + Ftf[row * DP + d] * Fte[row * DP + d];
        HP[row * P2 + d]      = (_Float16)ow;
        HP[row * P2 + D_ + d] = (_Float16)ie;
    }
    for (int i = tid; i < T_ * (P2 - 2 * D_); i += 256) {  // zero pad columns
        int t = i / (P2 - 2 * D_), c = 2 * D_ + (i % (P2 - 2 * D_));
        HP[((size_t)b * T_ + t) * P2 + c] = (_Float16)0.f;
    }
}

// ---------------------------------------------------------------------------
// Head: pred = H @ pred_w + b ; sigmoid ; tgt ; per-block BCE partial sums.
// ---------------------------------------------------------------------------
__global__ __launch_bounds__(256) void pred_kernel(
    const _Float16* __restrict__ HH, const float* __restrict__ predw,
    const float* __restrict__ predb, const int* __restrict__ target,
    float* __restrict__ out, float* __restrict__ partials)
{
    __shared__ float red[256];
    int i = blockIdx.x * 256 + threadIdx.x;
    float term = 0.f;
    if (i < M_) {
        float acc = predb[0];
        for (int k = 0; k < FC_; k++)
            acc += (float)HH[(size_t)i * FC_ + k] * predw[k];
        float sp = 1.f / (1.f + __expf(-acc));
        float tg = (float)(target[i] - 1);
        out[1 + i]      = sp;
        out[1 + M_ + i] = tg;
        term = fmaxf(acc, 0.f) - acc * tg + log1pf(__expf(-fabsf(acc)));
    }
    red[threadIdx.x] = term;
    __syncthreads();
    for (int o = 128; o > 0; o >>= 1) {
        if (threadIdx.x < o) red[threadIdx.x] += red[threadIdx.x + o];
        __syncthreads();
    }
    if (threadIdx.x == 0) partials[blockIdx.x] = red[0];
}

__global__ __launch_bounds__(128) void loss_finalize(
    const float* __restrict__ partials, int n, float* __restrict__ out)
{
    __shared__ float red[128];
    int tid = threadIdx.x;
    float s = 0.f;
    for (int i = tid; i < n; i += 128) s += partials[i];
    red[tid] = s;
    __syncthreads();
    for (int o = 64; o > 0; o >>= 1) {
        if (tid < o) red[tid] += red[tid + o];
        __syncthreads();
    }
    if (tid == 0) out[0] = red[0] / (float)M_;
}

// ---------------------------------------------------------------------------
extern "C" void kernel_launch(void* const* d_in, const int* in_sizes, int n_in,
                              void* d_out, int out_size, void* d_ws, size_t ws_size,
                              hipStream_t stream)
{
    (void)in_sizes; (void)n_in; (void)out_size; (void)ws_size;

    const int*   q_data     = (const int*)  d_in[0];
    const int*   qa_data    = (const int*)  d_in[1];
    const int*   target     = (const int*)  d_in[2];
    const int*   time_data  = (const int*)  d_in[3];
    const int*   att_data   = (const int*)  d_in[4];
    const int*   hint_data  = (const int*)  d_in[5];
    const int*   hintT_data = (const int*)  d_in[6];
    const float* q_embed    = (const float*)d_in[7];
    const float* qa_embed   = (const float*)d_in[8];
    const float* t_embed    = (const float*)d_in[9];
    const float* a_embed    = (const float*)d_in[10];
    const float* ht_embed   = (const float*)d_in[11];
    const float* diff_w     = (const float*)d_in[12];
    const float* diff_b     = (const float*)d_in[13];
    const float* diff2_w    = (const float*)d_in[14];
    const float* diff2_b    = (const float*)d_in[15];
    const float* keyw_w     = (const float*)d_in[16];
    const float* keyw_b     = (const float*)d_in[17];
    const float* valw_w     = (const float*)d_in[18];
    const float* valw_b     = (const float*)d_in[19];
    const float* erase_w    = (const float*)d_in[20];
    const float* erase_b    = (const float*)d_in[21];
    const float* add_w      = (const float*)d_in[22];
    const float* add_b      = (const float*)d_in[23];
    const float* ln_gamma   = (const float*)d_in[24];
    const float* ln_beta    = (const float*)d_in[25];
    const float* read_w     = (const float*)d_in[26];
    const float* read_b     = (const float*)d_in[27];
    const float* pred_w     = (const float*)d_in[28];
    const float* pred_b     = (const float*)d_in[29];
    const float* mem_key    = (const float*)d_in[30];
    const float* mem_value0 = (const float*)d_in[31];
    float* out = (float*)d_out;

    // ---- workspace bump allocator (256B aligned) ----
    char* w = (char*)d_ws;
    size_t off = 0;
    auto alloc = [&](size_t bytes) -> char* {
        char* p = w + off;
        off += bytes;
        off = (off + 255) & ~(size_t)255;
        return p;
    };

    _Float16* Wdiff  = (_Float16*)alloc((size_t)DP * DP * 2);
    _Float16* Wdiff2 = (_Float16*)alloc((size_t)DP * DP * 2);
    _Float16* Wkey   = (_Float16*)alloc((size_t)DP * DP * 2);
    _Float16* Wval   = (_Float16*)alloc((size_t)DP * DP * 2);
    _Float16* Werase = (_Float16*)alloc((size_t)DP * DP * 2);
    _Float16* Wadd   = (_Float16*)alloc((size_t)DP * DP * 2);
    _Float16* Wread  = (_Float16*)alloc((size_t)FC_ * P2 * 2);

    _Float16* HA  = (_Float16*)alloc((size_t)M_ * DP * 2);   // t_e+q_e, later Y
    _Float16* HB  = (_Float16*)alloc((size_t)M_ * DP * 2);   // ctx
    _Float16* HQA = (_Float16*)alloc((size_t)M_ * DP * 2);   // qa_e
    _Float16* HC  = (_Float16*)alloc((size_t)M_ * DP * 2);   // tanh hidden temp
    _Float16* HR  = (_Float16*)alloc((size_t)M_ * DP * 2);   // key reads
    _Float16* HVR = (_Float16*)alloc((size_t)M_ * DP * 2);   // value reads
    _Float16* HP  = (_Float16*)alloc((size_t)M_ * P2 * 2);   // predict_input
    _Float16* HH  = (_Float16*)alloc((size_t)M_ * FC_ * 2);  // tanh(read fc)

    float* Fqe  = (float*)alloc((size_t)M_ * DP * 4);
    float* Fte  = (float*)alloc((size_t)M_ * DP * 4);
    float* Fctx = (float*)alloc((size_t)M_ * DP * 4);
    float* Ftf  = (float*)alloc((size_t)M_ * DP * 4);   // time_final
    float* Fdf  = (float*)alloc((size_t)M_ * DP * 4);   // diff_final
    float* Fqf  = (float*)alloc((size_t)M_ * DP * 4);   // q_e final
    float* Fe   = (float*)alloc((size_t)M_ * DP * 4);   // erase
    float* Fa   = (float*)alloc((size_t)M_ * DP * 4);   // add
    float* Frv  = (float*)alloc((size_t)M_ * DP * 4);   // keyw(reads)
    float* Frvk = (float*)alloc((size_t)M_ * DP * 4);   // valw(vreads)
    float* partials = (float*)alloc(256 * 4);

    // ---- 1. pack weights to padded f16 ----
    {
        int n = DP * DP, g = (n + 255) / 256;
        pack_weight<<<g, 256, 0, stream>>>(diff_w,  Wdiff,  D_, D_, DP, DP);
        pack_weight<<<g, 256, 0, stream>>>(diff2_w, Wdiff2, D_, D_, DP, DP);
        pack_weight<<<g, 256, 0, stream>>>(keyw_w,  Wkey,   D_, D_, DP, DP);
        pack_weight<<<g, 256, 0, stream>>>(valw_w,  Wval,   D_, D_, DP, DP);
        pack_weight<<<g, 256, 0, stream>>>(erase_w, Werase, D_, D_, DP, DP);
        pack_weight<<<g, 256, 0, stream>>>(add_w,   Wadd,   D_, D_, DP, DP);
        int n2 = FC_ * P2;
        pack_weight<<<(n2 + 255) / 256, 256, 0, stream>>>(read_w, Wread, FC_, 2 * D_, P2, FC_);
    }

    // ---- 2. embeddings + operand packing ----
    embed_pack<<<M_, 224, 0, stream>>>(q_data, qa_data, time_data, att_data,
                                       hint_data, hintT_data,
                                       q_embed, qa_embed, t_embed, a_embed, ht_embed,
                                       HA, HB, HQA, Fqe, Fte, Fctx);

    // ---- 3. pre-scan GEMMs (WMMA) ----
    const int nT    = DP / 16;                 // 14
    const int tiles = (M_ / 16) * nT;          // 22400
    const int gD    = (tiles + 7) / 8;
    // time_final = sigmoid(diff2(tanh(diff(t_e+q_e))))
    gemm_wmma<<<gD, 256, 0, stream>>>(HA, Wdiff,  diff_b,  D_, nullptr, HC, DP, nT, tiles, DP, 1);
    gemm_wmma<<<gD, 256, 0, stream>>>(HC, Wdiff2, diff2_b, D_, Ftf, nullptr, DP, nT, tiles, DP, 2);
    // diff_final = sigmoid(diff2(tanh(diff(ctx))))
    gemm_wmma<<<gD, 256, 0, stream>>>(HB, Wdiff,  diff_b,  D_, nullptr, HC, DP, nT, tiles, DP, 1);
    gemm_wmma<<<gD, 256, 0, stream>>>(HC, Wdiff2, diff2_b, D_, Fdf, nullptr, DP, nT, tiles, DP, 2);
    // q_e = diff(q_e + diff_final*ctx)
    combine_y<<<M_, 224, 0, stream>>>(Fqe, Fdf, Fctx, HA);
    gemm_wmma<<<gD, 256, 0, stream>>>(HA, Wdiff, diff_b, D_, Fqf, nullptr, DP, nT, tiles, DP, 0);
    // erase/add hoisted out of the scan (depend only on qa_e)
    gemm_wmma<<<gD, 256, 0, stream>>>(HQA, Werase, erase_b, D_, Fe, nullptr, DP, nT, tiles, DP, 2);
    gemm_wmma<<<gD, 256, 0, stream>>>(HQA, Wadd,   add_b,   D_, Fa, nullptr, DP, nT, tiles, DP, 1);

    // ---- 4. recurrent memory scan (LDS-resident state) ----
    scan_kernel<<<B_, 256, 0, stream>>>(Fqf, Fe, Fa, mem_key, mem_value0, HR, HVR);

    // ---- 5. post-scan GEMMs ----
    gemm_wmma<<<gD, 256, 0, stream>>>(HR,  Wkey, keyw_b, D_, Frv,  nullptr, DP, nT, tiles, DP, 0);
    gemm_wmma<<<gD, 256, 0, stream>>>(HVR, Wval, valw_b, D_, Frvk, nullptr, DP, nT, tiles, DP, 0);

    // ---- 6. LayerNorm + gate + predict_input pack ----
    ln_predictpack<<<B_, 256, 0, stream>>>(Frv, Frvk, Fqf, Ftf, Fte, ln_gamma, ln_beta, HP);

    // ---- 7. FC layer [25600,416] x [512,416]^T (WMMA) ----
    const int nT2    = FC_ / 16;               // 32
    const int tiles2 = (M_ / 16) * nT2;        // 51200
    gemm_wmma<<<(tiles2 + 7) / 8, 256, 0, stream>>>(HP, Wread, read_b, FC_,
                                                    nullptr, HH, P2, nT2, tiles2, FC_, 1);

    // ---- 8. head + deterministic BCE reduction ----
    pred_kernel<<<M_ / 256, 256, 0, stream>>>(HH, pred_w, pred_b, target, out, partials);
    loss_finalize<<<1, 128, 0, stream>>>(partials, M_ / 256, out);
}
